// MultiheadSelfAttention_20091857010966
// MI455X (gfx1250) — matrix operand order repaired and checked by reading, not compiled, not run on hardware
//
#include <hip/hip_runtime.h>

// ---------------------------------------------------------------------------
// Problem constants (from reference): B=8, S=1024, E=768, H=12, P=64
// ---------------------------------------------------------------------------
#define BB 8
#define SS 1024
#define EE 768
#define HH 12
#define PP 64
#define HP (HH * PP)          // 768
#define RSCALE 0.03608439182435161f   // 1/sqrt(768)

typedef __attribute__((ext_vector_type(16))) __bf16 v16bf;
typedef __attribute__((ext_vector_type(8)))  __bf16 v8bf;
typedef __attribute__((ext_vector_type(8)))  float  v8f;

union FragBF { v16bf v; v8bf h[2]; };

__device__ __forceinline__ __bf16 to_bf16(float f) {
  union { float f; unsigned u; } x; x.f = f;
  unsigned r = x.u + 0x7FFFu + ((x.u >> 16) & 1u);   // round-to-nearest-even
  unsigned short hs = (unsigned short)(r >> 16);
  __bf16 o; __builtin_memcpy(&o, &hs, 2); return o;
}

// A fragment: 16(M) x 32(K) bf16, A row-major (K contiguous), stride lda elems.
// Lane l: row = l&15 ; K chunks {8*(l>>4) .. +7} and {+16 .. +23}.  (ISA 7.12.2)
__device__ __forceinline__ FragBF load_a_frag(const __bf16* A, int lda, int lane) {
  int m  = lane & 15;
  int kh = (lane >> 4) << 3;                 // 0 or 8
  const __bf16* p = A + m * lda + kh;
  FragBF f;
  f.h[0] = *(const v8bf*)(p);
  f.h[1] = *(const v8bf*)(p + 16);
  return f;
}

// B fragment: 32(K) x 16(N) bf16 sourced from Bt (N x K row-major, K contiguous).
// Lane l: col = l&15 ; K chunks {8*(l>>4) .. +7} and {+16 .. +23}.
__device__ __forceinline__ FragBF load_b_frag(const __bf16* Bt, int ldb, int lane) {
  int n  = lane & 15;
  int kh = (lane >> 4) << 3;                 // 0 or 8
  const __bf16* p = Bt + n * ldb + kh;
  FragBF f;
  f.h[0] = *(const v8bf*)(p);
  f.h[1] = *(const v8bf*)(p + 16);
  return f;
}

__device__ __forceinline__ v8f wmma_bf16(const FragBF& a, const FragBF& b, v8f c) {
  return __builtin_amdgcn_wmma_f32_16x16x32_bf16(false, a.v, false, b.v,
                                                 (short)0, c, false, false);
}

// C/D layout (ISA 7.12.2): lane l holds N = l&15 ; VGPR r holds M = r + 8*(l>>4)
__device__ __forceinline__ void store_c_f32(float* C, int ldc, v8f c, int lane) {
  int n = lane & 15, mr = (lane >> 4) << 3;
#pragma unroll
  for (int r = 0; r < 8; ++r) C[(mr + r) * ldc + n] = c[r];
}
__device__ __forceinline__ void store_c_bf16(__bf16* C, int ldc, v8f c, int lane) {
  int n = lane & 15, mr = (lane >> 4) << 3;
#pragma unroll
  for (int r = 0; r < 8; ++r) C[(mr + r) * ldc + n] = to_bf16(c[r]);
}

// ---------------------------------------------------------------------------
// 32(M) x 64(N) wave tile, TN: A (32 x K row-major), Bt (64 x K row-major).
// acc[ni] = subtile (mi=0, ni), acc[4+ni] = subtile (mi=1, ni).
// Ping-pong pipelined: two fixed fragment sets sA/sB (no register rotation,
// so no v_mov churn).  While one set's 8 WMMAs run, the other set's 12 b128
// loads are in flight.  Requires K % 64 == 0 (all call sites: 768 or 1024).
// ---------------------------------------------------------------------------
struct FragSet { FragBF a0, a1, b0, b1, b2, b3; };

__device__ __forceinline__ FragSet load_set(const __bf16* A, int lda,
                                            const __bf16* Bt, int ldb,
                                            int k, int lane) {
  FragSet s;
  s.a0 = load_a_frag(A + k,            lda, lane);
  s.a1 = load_a_frag(A + 16 * lda + k, lda, lane);
  s.b0 = load_b_frag(Bt + k,                    ldb, lane);
  s.b1 = load_b_frag(Bt + (size_t)16 * ldb + k, ldb, lane);
  s.b2 = load_b_frag(Bt + (size_t)32 * ldb + k, ldb, lane);
  s.b3 = load_b_frag(Bt + (size_t)48 * ldb + k, ldb, lane);
  return s;
}

__device__ __forceinline__ void compute_set(const FragSet& s, v8f acc[8]) {
  acc[0] = wmma_bf16(s.a0, s.b0, acc[0]);
  acc[4] = wmma_bf16(s.a1, s.b0, acc[4]);
  acc[1] = wmma_bf16(s.a0, s.b1, acc[1]);
  acc[5] = wmma_bf16(s.a1, s.b1, acc[5]);
  acc[2] = wmma_bf16(s.a0, s.b2, acc[2]);
  acc[6] = wmma_bf16(s.a1, s.b2, acc[6]);
  acc[3] = wmma_bf16(s.a0, s.b3, acc[3]);
  acc[7] = wmma_bf16(s.a1, s.b3, acc[7]);
}

__device__ __forceinline__ void gemm_tile_32x64(const __bf16* A, int lda,
                                                const __bf16* Bt, int ldb,
                                                int K, int lane, v8f acc[8]) {
  FragSet sA = load_set(A, lda, Bt, ldb, 0,  lane);
  FragSet sB = load_set(A, lda, Bt, ldb, 32, lane);
#pragma unroll 1
  for (int k = 0; k < K - 64; k += 64) {
    compute_set(sA, acc);                        // step k
    sA = load_set(A, lda, Bt, ldb, k + 64, lane);
    compute_set(sB, acc);                        // step k+32
    sB = load_set(A, lda, Bt, ldb, k + 96, lane);
  }
  compute_set(sA, acc);                          // step K-64
  compute_set(sB, acc);                          // step K-32
}

// ---------------------------------------------------------------------------
// Conversion / transpose kernels (fp32 -> bf16 staging)
// ---------------------------------------------------------------------------
__global__ void cvt_f32_to_bf16(const float* __restrict__ in,
                                __bf16* __restrict__ out, int n) {
  int i = blockIdx.x * blockDim.x + threadIdx.x;
  int st = gridDim.x * blockDim.x;
  for (; i < n; i += st) out[i] = to_bf16(in[i]);
}

// kpT[h][p][e] = kp[h][e][p] (and same for vp)
__global__ void cvt_proj_T(const float* __restrict__ kp, const float* __restrict__ vp,
                           __bf16* __restrict__ kpT, __bf16* __restrict__ vpT) {
  int n = HH * PP * EE;
  int i = blockIdx.x * blockDim.x + threadIdx.x;
  int st = gridDim.x * blockDim.x;
  for (; i < n; i += st) {
    int e = i % EE; int t = i / EE; int p = t % PP; int h = t / PP;
    int src = (h * EE + e) * PP + p;
    kpT[i] = to_bf16(kp[src]);
    vpT[i] = to_bf16(vp[src]);
  }
}

// liftT[e][hp] = lifting[hp][e]
__global__ void cvt_lift_T(const float* __restrict__ lift, __bf16* __restrict__ liftT) {
  int n = EE * HP;
  int i = blockIdx.x * blockDim.x + threadIdx.x;
  int st = gridDim.x * blockDim.x;
  for (; i < n; i += st) {
    int hp = i % HP; int e = i / HP;
    liftT[i] = to_bf16(lift[hp * EE + e]);
  }
}

// ---------------------------------------------------------------------------
// Stage 1: k = x @ key_proj (natural), v = x @ value_proj (stored transposed)
// Block = (b, mt). All 24 jobs (12 heads x {k,v}) share the SAME A tile
// (x[b] rows mt*32..+31)  -> A lines served from WGP$ after the first wave.
// Each wave computes a full 32(M) x 64(N=P) tile.
// ---------------------------------------------------------------------------
__global__ void __launch_bounds__(256)
gemm_kv_kernel(const __bf16* __restrict__ xb,
               const __bf16* __restrict__ kpT, const __bf16* __restrict__ vpT,
               __bf16* __restrict__ kbuf, __bf16* __restrict__ vT) {
  int wave = threadIdx.x >> 5, lane = threadIdx.x & 31;
  int b  = blockIdx.x >> 5;                    // 0..7
  int mt = blockIdx.x & 31;                    // 0..31 row tiles of 32
  int hf = lane >> 4, col = lane & 15;
  const __bf16* A = xb + (size_t)b * SS * EE + (size_t)mt * 32 * EE;
  for (int job = wave; job < 2 * HH; job += 8) {
    int h = job >> 1;
    bool isV = job & 1;
    const __bf16* Bt = (isV ? vpT : kpT) + (size_t)h * PP * EE;
    v8f acc[8] = {};
    gemm_tile_32x64(A, EE, Bt, EE, EE, lane, acc);
    size_t bh = (size_t)(b * HH + h);
    if (!isV) {
      __bf16* C = kbuf + bh * SS * PP + (size_t)mt * 32 * PP;
#pragma unroll
      for (int mi = 0; mi < 2; ++mi)
#pragma unroll
        for (int ni = 0; ni < 4; ++ni)
          store_c_bf16(C + mi * 16 * PP + ni * 16, PP, acc[4 * mi + ni], lane);
    } else {
      __bf16* Vb = vT + bh * PP * SS;
#pragma unroll
      for (int mi = 0; mi < 2; ++mi)
#pragma unroll
        for (int ni = 0; ni < 4; ++ni) {
          v8f c = acc[4 * mi + ni];
#pragma unroll
          for (int r = 0; r < 8; ++r) {
            int srow = mt * 32 + mi * 16 + r + 8 * hf;
            int pcol = ni * 16 + col;
            Vb[(size_t)pcol * SS + srow] = to_bf16(c[r]);
          }
        }
    }
  }
}

// ---------------------------------------------------------------------------
// Stage 2: qv = q_heads @ v.
// Block = (h, mt); the 8 waves are the 8 batches b, all sharing the SAME
// A tile (q_heads[h] rows mt*32..+31, K=S) -> q streams from L2 once, not 8x.
// Bt = vT[b,h] (P x S). Wave tile 32 x 64 (full N=P).
// ---------------------------------------------------------------------------
__global__ void __launch_bounds__(256)
gemm_qv_kernel(const __bf16* __restrict__ qb, const __bf16* __restrict__ vT,
               __bf16* __restrict__ qvbuf) {
  int wave = threadIdx.x >> 5, lane = threadIdx.x & 31;
  int h  = blockIdx.x >> 5;                    // 0..11
  int mt = blockIdx.x & 31;                    // 0..31
  int b  = wave;                               // 8 waves = 8 batches
  const __bf16* A  = qb + (size_t)h * SS * SS + (size_t)mt * 32 * SS;
  size_t bh = (size_t)(b * HH + h);
  const __bf16* Bt = vT + bh * PP * SS;
  v8f acc[8] = {};
  gemm_tile_32x64(A, SS, Bt, SS, SS, lane, acc);
  __bf16* C = qvbuf + bh * SS * PP + (size_t)mt * 32 * PP;
#pragma unroll
  for (int mi = 0; mi < 2; ++mi)
#pragma unroll
    for (int ni = 0; ni < 4; ++ni)
      store_c_bf16(C + mi * 16 * PP + ni * 16, PP, acc[4 * mi + ni], lane);
}

// ---------------------------------------------------------------------------
// Stage 3-5 fused: flash attention.
// scores[i,j] = (k[i,:] . qv[j,:]) / sqrt(E), causal mask j<=i, softmax,
// attn[i,:]  = probs @ v  -> written to attn[b][i][h*P + p] (bf16).
// One wave owns a 16-row strip; probs pass through LDS for C->A relayout.
// Consecutive waves in a block walk the same (b,h) qv/vT streams -> WGP$ reuse.
// ---------------------------------------------------------------------------
#define PST 40   // LDS row stride (elems) for the 16x32 probs tile
__global__ void __launch_bounds__(256)
fa_kernel(const __bf16* __restrict__ kbuf, const __bf16* __restrict__ qvbuf,
          const __bf16* __restrict__ vT, __bf16* __restrict__ attn) {
  __shared__ __bf16 psh[8][16 * PST];
  int wave = threadIdx.x >> 5, lane = threadIdx.x & 31;
  int gw = blockIdx.x * 8 + wave;               // 0 .. B*H*64-1
  int strip = gw & 63;
  int bh = gw >> 6;
  if (bh >= BB * HH) return;
  int b = bh / HH, h = bh % HH;
  int hf = lane >> 4, col = lane & 15;
  const __bf16* kb = kbuf  + (size_t)bh * SS * PP;
  const __bf16* qv = qvbuf + (size_t)bh * SS * PP;
  const __bf16* vb = vT    + (size_t)bh * PP * SS;
  int i0 = strip * 16;

  FragBF qa0 = load_a_frag(kb + (size_t)i0 * PP + 0,  PP, lane);
  FragBF qa1 = load_a_frag(kb + (size_t)i0 * PP + 32, PP, lane);

  v8f acc[4] = {};
  float mi[8], li[8];
#pragma unroll
  for (int r = 0; r < 8; ++r) { mi[r] = -3.0e38f; li[r] = 0.f; }

  int nj = i0 / 32 + 1;                          // j-blocks up to the diagonal
#pragma unroll 1
  for (int jb = 0; jb < nj; ++jb) {
    int j0 = jb * 32;
    v8f sacc0 = {}, sacc1 = {};
    {
      FragBF b00 = load_b_frag(qv + (size_t)(j0 + 0)  * PP + 0,  PP, lane);
      FragBF b01 = load_b_frag(qv + (size_t)(j0 + 0)  * PP + 32, PP, lane);
      FragBF b10 = load_b_frag(qv + (size_t)(j0 + 16) * PP + 0,  PP, lane);
      FragBF b11 = load_b_frag(qv + (size_t)(j0 + 16) * PP + 32, PP, lane);
      sacc0 = wmma_bf16(qa0, b00, sacc0);
      sacc0 = wmma_bf16(qa1, b01, sacc0);
      sacc1 = wmma_bf16(qa0, b10, sacc1);
      sacc1 = wmma_bf16(qa1, b11, sacc1);
    }
    // scale + causal mask (C layout: row = r + 8*hf, col = nt*16 + (lane&15))
    float sv0[8], sv1[8];
#pragma unroll
    for (int r = 0; r < 8; ++r) {
      int rowg = i0 + r + 8 * hf;
      float a = sacc0[r] * RSCALE;
      float c = sacc1[r] * RSCALE;
      sv0[r] = ((j0 + col)      <= rowg) ? a : -3.0e38f;
      sv1[r] = ((j0 + 16 + col) <= rowg) ? c : -3.0e38f;
    }
    // online softmax per row (reduce across the 16 lanes sharing hf)
#pragma unroll
    for (int r = 0; r < 8; ++r) {
      float rm = fmaxf(sv0[r], sv1[r]);
#pragma unroll
      for (int off = 1; off < 16; off <<= 1) rm = fmaxf(rm, __shfl_xor(rm, off, 32));
      float mn = fmaxf(mi[r], rm);
      float scale = (mi[r] < -1.0e37f) ? 0.f : __expf(mi[r] - mn);
      float p0 = (sv0[r] < -1.0e37f) ? 0.f : __expf(sv0[r] - mn);
      float p1 = (sv1[r] < -1.0e37f) ? 0.f : __expf(sv1[r] - mn);
      psh[wave][(r + 8 * hf) * PST + col]      = to_bf16(p0);
      psh[wave][(r + 8 * hf) * PST + 16 + col] = to_bf16(p1);
      float rs = p0 + p1;
#pragma unroll
      for (int off = 1; off < 16; off <<= 1) rs += __shfl_xor(rs, off, 32);
      li[r] = li[r] * scale + rs;
      mi[r] = mn;
      acc[0][r] *= scale; acc[1][r] *= scale;
      acc[2][r] *= scale; acc[3][r] *= scale;
    }
    __builtin_amdgcn_wave_barrier();             // order ds stores before reloads
    // probs back as an A fragment (16 x 32 over j)
    FragBF pa = load_a_frag(&psh[wave][0], PST, lane);
    // O += P @ V  : Bt = vT rows p, K = j0..j0+31
#pragma unroll
    for (int nt2 = 0; nt2 < 4; ++nt2) {
      FragBF bv = load_b_frag(vb + (size_t)(nt2 * 16) * SS + j0, SS, lane);
      acc[nt2] = wmma_bf16(pa, bv, acc[nt2]);
    }
  }
  // epilogue: normalize and write attn[b][i][h*P + p]
#pragma unroll
  for (int nt2 = 0; nt2 < 4; ++nt2)
#pragma unroll
    for (int r = 0; r < 8; ++r) {
      float inv = (li[r] > 0.f) ? 1.f / li[r] : 0.f;
      int rowg = i0 + r + 8 * hf;
      int pcol = h * PP + nt2 * 16 + col;
      attn[((size_t)b * SS + rowg) * HP + pcol] = to_bf16(acc[nt2][r] * inv);
    }
}

// ---------------------------------------------------------------------------
// Stage 6: out = attn (B*S x HP) @ lifting (HP x E), via liftT (E x HP).
// Block = mt; all 12 n-blocks of 64 share the same attn row tile via WGP$.
// ---------------------------------------------------------------------------
__global__ void __launch_bounds__(256)
gemm_final_kernel(const __bf16* __restrict__ attn, const __bf16* __restrict__ liftT,
                  float* __restrict__ out) {
  int wave = threadIdx.x >> 5, lane = threadIdx.x & 31;
  int mt = blockIdx.x;                          // 0..255
  const __bf16* A = attn + (size_t)mt * 32 * HP;
  for (int nb = wave; nb < EE / 64; nb += 8) {  // 12 n-blocks of 64
    const __bf16* Bt = liftT + (size_t)nb * 64 * HP;
    v8f acc[8] = {};
    gemm_tile_32x64(A, HP, Bt, HP, HP, lane, acc);
    float* C = out + (size_t)mt * 32 * EE + (size_t)nb * 64;
#pragma unroll
    for (int mi = 0; mi < 2; ++mi)
#pragma unroll
      for (int ni = 0; ni < 4; ++ni)
        store_c_f32(C + mi * 16 * EE + ni * 16, EE, acc[4 * mi + ni], lane);
  }
}

// ---------------------------------------------------------------------------
extern "C" void kernel_launch(void* const* d_in, const int* in_sizes, int n_in,
                              void* d_out, int out_size, void* d_ws, size_t ws_size,
                              hipStream_t stream) {
  const float* x    = (const float*)d_in[0];   // (B,1,S,E)
  const float* kp   = (const float*)d_in[1];   // (H,E,P)
  const float* vp   = (const float*)d_in[2];   // (H,E,P)
  const float* q    = (const float*)d_in[3];   // (H,S,S)
  const float* lift = (const float*)d_in[4];   // (1,HP,E)
  float* out = (float*)d_out;

  char* ws = (char*)d_ws;
  size_t off = 0;
  auto alloc = [&](size_t bytes) -> void* {
    void* p = ws + off;
    off = (off + bytes + 255) & ~(size_t)255;
    return p;
  };
  __bf16* xb    = (__bf16*)alloc((size_t)BB * SS * EE * 2);
  __bf16* kpT   = (__bf16*)alloc((size_t)HH * PP * EE * 2);
  __bf16* vpT   = (__bf16*)alloc((size_t)HH * PP * EE * 2);
  __bf16* qb    = (__bf16*)alloc((size_t)HH * SS * SS * 2);
  __bf16* liftT = (__bf16*)alloc((size_t)EE * HP * 2);
  __bf16* kbuf  = (__bf16*)alloc((size_t)BB * HH * SS * PP * 2);
  __bf16* qvbuf = (__bf16*)alloc((size_t)BB * HH * SS * PP * 2);
  __bf16* vT    = (__bf16*)alloc((size_t)BB * HH * PP * SS * 2);
  __bf16* attn  = (__bf16*)alloc((size_t)BB * SS * HP * 2);
  (void)ws_size; (void)in_sizes; (void)n_in; (void)out_size;

  // fp32 -> bf16 staging (+ operand transposes for TN fragment layout)
  cvt_f32_to_bf16<<<4096, 256, 0, stream>>>(x, xb, BB * SS * EE);
  cvt_f32_to_bf16<<<8192, 256, 0, stream>>>(q, qb, HH * SS * SS);
  cvt_proj_T<<<576, 256, 0, stream>>>(kp, vp, kpT, vpT);
  cvt_lift_T<<<576, 256, 0, stream>>>(lift, liftT);

  // k (natural) and v (transposed) projections: block = (b, mt)
  gemm_kv_kernel<<<BB * 32, 256, 0, stream>>>(xb, kpT, vpT, kbuf, vT);
  // qv = q_heads @ v: block = (h, mt), waves = batches
  gemm_qv_kernel<<<HH * 32, 256, 0, stream>>>(qb, vT, qvbuf);
  // fused causal softmax(k @ qv^T / sqrt(E)) @ v  -> attn (B,S,HP)
  fa_kernel<<<768, 256, 0, stream>>>(kbuf, qvbuf, vT, attn);
  // out = attn @ lifting
  gemm_final_kernel<<<(BB * SS) / 32, 256, 0, stream>>>(attn, liftT, out);
}
